// SimpleLSTM_30107720745142
// MI455X (gfx1250) — matrix operand order repaired and checked by reading, not compiled, not run on hardware
//
#include <hip/hip_runtime.h>
#include <hip/hip_bf16.h>

#define HID 1024
#define NB 64          // batch
#define TT 512         // timesteps
#define KDIM 2048      // HID + 1024
#define GOUT 4096      // 4*HID

typedef __attribute__((ext_vector_type(16))) __bf16 v16bf;
typedef __attribute__((ext_vector_type(8)))  float  v8f;
typedef __attribute__((ext_vector_type(4)))  float  v4f;

__device__ __forceinline__ float sigmoidf_(float x) {
    return 1.0f / (1.0f + __expf(-x));
}

// WGP-scope prefetch (ISA 10.5: SCOPE 0 pulls the line into ALL cache levels
// incl. WGP$). __builtin_prefetch lowers to scope:SCOPE_SE, which skips the
// WGP cache and is useless for L2-resident data, so use inline asm.
__device__ __forceinline__ void prefetch_wgp(const void* p) {
    asm volatile("global_prefetch_b8 %0, off" :: "v"(p));
}

// Convert 16 f32 (A-fragment order) -> v16bf; lowers to v_cvt_pk_bf16_f32.
__device__ __forceinline__ v16bf cvt_afrag(const v4f* a) {
    v16bf out;
    const float* f = reinterpret_cast<const float*>(a);
#pragma unroll
    for (int e = 0; e < 16; ++e) out[e] = (__bf16)f[e];
    return out;
}

// ---------------------------------------------------------------------------
// One-time: W (4096 x 2048, f32 row-major) -> bf16 row-major in scratch.
// W fits easily in the 192MB L2 (16MB as bf16), so per-step B traffic is
// L2-resident.
// ---------------------------------------------------------------------------
__global__ void convert_w_bf16(const float* __restrict__ W,
                               __bf16* __restrict__ Wbf) {
    const int total = GOUT * KDIM;
    for (int i = blockIdx.x * blockDim.x + threadIdx.x; i < total;
         i += gridDim.x * blockDim.x) {
        Wbf[i] = (__bf16)W[i];
    }
}

// ---------------------------------------------------------------------------
// Per step: fused pack + GEMM (64x2048 @ 2048x4096) + bias + gates + update.
// Grid: 256 blocks, block = 256 threads = 8 waves (wave32).
//   blockIdx.x = m + 4*j : m in 0..3 (16-row batch tile), j in 0..63
//                          (16 hidden cols; owns gate cols j*16 + {0..3}*1024)
// Wave w takes K slice [w*256, w*256+256):
//   waves 0..3  -> A source = h_{t-1} (H_all[:,t-1,:], f32; zero at t==0)
//   waves 4..7  -> A source = Emb[X[:,t]]            (f32)
// A is loaded as f32 and converted to bf16 in registers (co-executes with
// XDL WMMA).
//
// The backend serializes each demand load against its WMMA
// (s_wait_loadcnt 0x0), so latency is hidden with WGP-scope
// global_prefetch_b8 one 128B cacheline ahead of each stream: prefetches
// have no counter and no dest VGPRs, so the scheduler can't sink or
// collapse them, and the demand loads then hit the WGP cache instead of L2.
// B rows advance 64B/chunk (one line per 2 chunks -> prefetch on even kc);
// the A row advances 128B/chunk (prefetch every kc).
// Partials reduced via 32KB LDS; wave r does pointwise for C-register row r.
// ---------------------------------------------------------------------------
__global__ void
__attribute__((amdgpu_flat_work_group_size(256, 256)))
__attribute__((amdgpu_waves_per_eu(1, 2)))
lstm_step_kernel(const __bf16* __restrict__ Wbf,  // [4096][2048] bf16
                 const float* __restrict__ Hall,  // [64][512][1024] (in)
                 const float* __restrict__ Emb,   // [32000][1024]
                 const int*   __restrict__ X,     // [64][512]
                 const float* __restrict__ bias,  // [4096]
                 float* __restrict__ c_state,     // [64][1024]
                 float* __restrict__ Hout,        // [64][512][1024] (out)
                 int t) {
    __shared__ float red[8][4][8][32];   // [wave][gate][reg][lane] = 32 KB

    const int tid   = threadIdx.x;
    const int wave  = tid >> 5;
    const int lane  = tid & 31;
    const int row16 = lane & 15;
    const int half  = lane >> 4;

    const int m = blockIdx.x & 3;     // batch tile
    const int j = blockIdx.x >> 2;    // hidden tile

    const int arow = m * 16 + row16;  // batch row this lane's A fragment uses

    // B fragment (32x16 bf16), B[k][n] = W[g*1024 + j*16 + n][k]:
    //   lane holds N=row16; elems 0..7 -> K = k0 + half*8 + e, elems 8..15 -> K = k0 + 16 + half*8 + (e-8)
    const __bf16* Bp[4];
#pragma unroll
    for (int g = 0; g < 4; ++g)
        Bp[g] = Wbf + (size_t)(g * HID + j * 16 + row16) * KDIM;

    v8f acc0 = {}, acc1 = {}, acc2 = {}, acc3 = {};

    const bool a_is_zero = (wave < 4) && (t == 0);

    if (!a_is_zero) {
        // A source pointer for this wave's 256-wide K slice (f32).
        const float* srcA;
        if (wave < 4) {
            srcA = Hall + ((size_t)arow * TT + (t - 1)) * HID + wave * 256;
        } else {
            const int tok = X[arow * TT + t];
            srcA = Emb + (size_t)tok * HID + (wave * 256 - HID);
        }

        // Warm WGP$ for the first line of each stream.
        prefetch_wgp(Bp[0] + half * 16);
        prefetch_wgp(Bp[1] + half * 16);
        prefetch_wgp(Bp[2] + half * 16);
        prefetch_wgp(Bp[3] + half * 16);
        prefetch_wgp(srcA + half * 8);
        prefetch_wgp(srcA + 32 + half * 8);

        // A fragment (16x32 bf16): lane holds M=arow,
        //   elems 0..7  -> K = k0 + half*8 + {0..7}
        //   elems 8..15 -> K = k0 + half*8 + 16 + {0..7}
        auto loadA = [&](v4f* dst, int kc) {
            const float* s = srcA + kc * 32 + half * 8;
            dst[0] = *reinterpret_cast<const v4f*>(s);
            dst[1] = *reinterpret_cast<const v4f*>(s + 4);
            dst[2] = *reinterpret_cast<const v4f*>(s + 16);
            dst[3] = *reinterpret_cast<const v4f*>(s + 20);
        };
        auto loadB = [&](v16bf* dst, int kc) {
            const int off = kc * 32 + half * 8;
#pragma unroll
            for (int g = 0; g < 4; ++g) {
                v4f* p = reinterpret_cast<v4f*>(&dst[g]);
                p[0] = *reinterpret_cast<const v4f*>(Bp[g] + off);
                p[1] = *reinterpret_cast<const v4f*>(Bp[g] + off + 16);
            }
        };

        v4f   a_cur[4], a_nxt[4];
        v16bf b_cur[4], b_nxt[4];
        loadA(a_cur, 0);
        loadB(b_cur, 0);

#pragma unroll
        for (int kc = 0; kc < 8; ++kc) {
            if (kc < 6) {
                if ((kc & 1) == 0) {
                    // Next 128B line of each B row (covers chunks kc+2, kc+3).
                    const int poff = (kc + 2) * 32;
                    prefetch_wgp(Bp[0] + poff);
                    prefetch_wgp(Bp[1] + poff);
                    prefetch_wgp(Bp[2] + poff);
                    prefetch_wgp(Bp[3] + poff);
                }
                // Next 128B line of the A row.
                prefetch_wgp(srcA + (kc + 2) * 32 + half * 8);
            }
            if (kc < 7) {               // demand loads for next chunk
                loadA(a_nxt, kc + 1);
                loadB(b_nxt, kc + 1);
            }

            const v16bf afrag = cvt_afrag(a_cur);

            acc0 = __builtin_amdgcn_wmma_f32_16x16x32_bf16(false, afrag, false, b_cur[0],
                                                           (short)0, acc0, false, false);
            acc1 = __builtin_amdgcn_wmma_f32_16x16x32_bf16(false, afrag, false, b_cur[1],
                                                           (short)0, acc1, false, false);
            acc2 = __builtin_amdgcn_wmma_f32_16x16x32_bf16(false, afrag, false, b_cur[2],
                                                           (short)0, acc2, false, false);
            acc3 = __builtin_amdgcn_wmma_f32_16x16x32_bf16(false, afrag, false, b_cur[3],
                                                           (short)0, acc3, false, false);

            if (kc < 7) {
#pragma unroll
                for (int q = 0; q < 4; ++q) { a_cur[q] = a_nxt[q]; b_cur[q] = b_nxt[q]; }
            }
        }
    }
    // else: h == 0 at t==0, partial products are exactly zero; acc stays {}.

#pragma unroll
    for (int r = 0; r < 8; ++r) {
        red[wave][0][r][lane] = acc0[r];
        red[wave][1][r][lane] = acc1[r];
        red[wave][2][r][lane] = acc2[r];
        red[wave][3][r][lane] = acc3[r];
    }
    __syncthreads();

    // Pointwise: wave 'r' handles C-register row r.
    // C/D layout: reg r, lanes 0-15 -> M=r, lanes 16-31 -> M=8+r; N = lane%16.
    {
        const int r   = wave;
        const int row = m * 16 + r + half * 8;   // batch row
        const int col = j * 16 + row16;          // hidden col

        float gv[4];
#pragma unroll
        for (int g = 0; g < 4; ++g) {
            float s = bias[g * HID + col];
#pragma unroll
            for (int w = 0; w < 8; ++w) s += red[w][g][r][lane];
            gv[g] = s;
        }

        const float fg = sigmoidf_(gv[0]);
        const float ig = sigmoidf_(gv[1]);
        const float og = sigmoidf_(gv[2]);
        const float gh = tanhf(gv[3]);

        const float cold = (t == 0) ? 0.0f : c_state[row * HID + col];
        const float cnew = fg * cold + ig * gh;
        c_state[row * HID + col] = cnew;

        const float h = og * tanhf(cnew);
        Hout[((size_t)row * TT + t) * HID + col] = h;
    }
}

// ---------------------------------------------------------------------------
// H_last = H_all[:, T-1, :]
// ---------------------------------------------------------------------------
__global__ void copy_hlast(const float* __restrict__ Hall,
                           float* __restrict__ Hlast) {
    int idx = blockIdx.x * blockDim.x + threadIdx.x;   // 0 .. 64*1024-1
    if (idx >= NB * HID) return;
    int row = idx >> 10;
    int col = idx & (HID - 1);
    Hlast[idx] = Hall[((size_t)row * TT + (TT - 1)) * HID + col];
}

// ---------------------------------------------------------------------------
// launch
// ---------------------------------------------------------------------------
extern "C" void kernel_launch(void* const* d_in, const int* in_sizes, int n_in,
                              void* d_out, int out_size, void* d_ws, size_t ws_size,
                              hipStream_t stream) {
    (void)in_sizes; (void)n_in; (void)out_size; (void)ws_size;

    const int*   X   = (const int*)d_in[0];     // (64, 512) token ids
    const float* Emb = (const float*)d_in[1];   // (32000, 1024)
    const float* W   = (const float*)d_in[2];   // (4096, 2048)
    const float* b   = (const float*)d_in[3];   // (4096,)

    float* Hall  = (float*)d_out;               // (64, 512, 1024); h state lives here
    float* Hlast = Hall + (size_t)NB * TT * HID;

    // Scratch (~16.25 MB): Wbf (4096*2048 bf16) + c_state (64*1024 f32)
    __bf16* Wbf = (__bf16*)d_ws;
    float* c_state = (float*)(Wbf + (size_t)GOUT * KDIM);

    convert_w_bf16<<<1024, 256, 0, stream>>>(W, Wbf);

    for (int t = 0; t < TT; ++t) {
        lstm_step_kernel<<<256, 256, 0, stream>>>(Wbf, Hall, Emb, X, b,
                                                  c_state, Hall, t);
    }

    copy_hlast<<<(NB * HID + 255) / 256, 256, 0, stream>>>(Hall, Hlast);
}
